// MyAttentionBlock_37400575213666
// MI455X (gfx1250) — matrix operand order repaired and checked by reading, not compiled, not run on hardware
//
#include <hip/hip_runtime.h>

// ---------------- types ----------------
typedef __attribute__((ext_vector_type(16))) __bf16          v16bf;
typedef __attribute__((ext_vector_type(8)))  float           v8f;
typedef __attribute__((ext_vector_type(16))) unsigned short  v16u;
typedef __attribute__((ext_vector_type(8)))  unsigned short  v8u;
typedef __attribute__((ext_vector_type(4)))  unsigned int    u32x4;
typedef __attribute__((ext_vector_type(8)))  int             i32x8;
typedef __attribute__((ext_vector_type(4)))  int             i32x4;

#define S_TOT   4096
#define DIMK    1024
#define N_QKV   3072
#define NHEAD   16
#define HDIM    64

__device__ __forceinline__ unsigned short f2bf(float f) {
    return __builtin_bit_cast(unsigned short, (__bf16)f);
}
__device__ __forceinline__ unsigned f2bf2(float x, float y) {
    return (unsigned)f2bf(x) | ((unsigned)f2bf(y) << 16);
}

__device__ __forceinline__ v8f wmma_bf16(v16u a, v16u b, v8f c) {
    return __builtin_amdgcn_wmma_f32_16x16x32_bf16(
        false, __builtin_bit_cast(v16bf, a),
        false, __builtin_bit_cast(v16bf, b),
        (short)0, c, false, false);
}

__device__ __forceinline__ v16u ld2(const unsigned short* p0, const unsigned short* p1) {
    v8u a = *(const v8u*)p0;
    v8u b = *(const v8u*)p1;
    return __builtin_shufflevector(a, b, 0,1,2,3,4,5,6,7,8,9,10,11,12,13,14,15);
}

// =====================================================================
// Kernel 0: one-shot fp32 -> bf16 conversion (x, w_qkv, w_proj)
// =====================================================================
__global__ __launch_bounds__(256) void cvt_bf16(
        const float4* __restrict__ src, uint2* __restrict__ dst, int n4) {
    int i = blockIdx.x * 256 + threadIdx.x;
    if (i < n4) {
        float4 v = src[i];
        uint2 o;
        o.x = f2bf2(v.x, v.y);
        o.y = f2bf2(v.z, v.w);
        dst[i] = o;
    }
}

// =====================================================================
// Kernel 1: qkv = x @ w_qkv + b ; scatter Q/K/V (bf16) head-major
// =====================================================================
__global__ __launch_bounds__(256) void qkv_gemm(
        const unsigned short* __restrict__ xb, const unsigned short* __restrict__ wb,
        const float* __restrict__ bias,
        unsigned short* __restrict__ q_ws, unsigned short* __restrict__ k_ws,
        unsigned short* __restrict__ v_ws) {
    __shared__ unsigned short alds[64][40];    // [m][k]
    __shared__ unsigned short blds[128][40];   // [n][k]

    const int tid  = threadIdx.x;
    const int wave = tid >> 5, lane = tid & 31;
    const int wm   = wave & 3, wn = wave >> 2;
    const int half = lane >> 4, l16 = lane & 15;
    const int m0   = blockIdx.x * 64;
    const int n0   = blockIdx.y * 128;

    v8f acc[4] = {};

    for (int k0 = 0; k0 < DIMK; k0 += 32) {
        {   // stage A: 64x32 bf16, one b128 copy per thread
            int r  = tid >> 2;
            int c8 = (tid & 3) * 8;
            *(v8u*)&alds[r][c8] = *(const v8u*)(xb + (size_t)(m0 + r) * DIMK + k0 + c8);
        }
        {   // stage B: 32x128 bf16, transposed to [n][k]
            int r  = tid >> 4;
            int c8 = (tid & 15) * 8;
            #pragma unroll
            for (int rr = 0; rr < 32; rr += 16) {
                v8u v = *(const v8u*)(wb + (size_t)(k0 + r + rr) * N_QKV + n0 + c8);
                #pragma unroll
                for (int c = 0; c < 8; ++c) blds[c8 + c][r + rr] = v[c];
            }
        }
        __syncthreads();

        const unsigned short* arow = &alds[wm * 16 + l16][0];
        v16u af = ld2(arow + half * 8, arow + 16 + half * 8);
        v16u bfr[4];
        #pragma unroll
        for (int t = 0; t < 4; ++t) {
            const unsigned short* brow = &blds[wn * 64 + t * 16 + l16][0];
            bfr[t] = ld2(brow + half * 8, brow + 16 + half * 8);
        }
        #pragma unroll
        for (int t = 0; t < 4; ++t) acc[t] = wmma_bf16(af, bfr[t], acc[t]);
        __syncthreads();
    }

    // epilogue: tile base is wave-uniform -> force scalar branches
    #pragma unroll
    for (int t = 0; t < 4; ++t) {
        const int nb = __builtin_amdgcn_readfirstlane(n0 + wn * 64 + t * 16);
        float bv = bias[nb + l16];
        unsigned short* dst;
        int nn = nb;
        if (nn < 1024)      { dst = q_ws; }
        else if (nn < 2048) { dst = k_ws; nn -= 1024; }
        else                { dst = v_ws; nn -= 2048; }
        const int head = nn >> 6;
        unsigned short* base = dst +
            ((size_t)head * S_TOT + m0 + wm * 16 + half * 8) * HDIM + (nn & 63) + l16;
        #pragma unroll
        for (int e = 0; e < 8; ++e)
            base[(size_t)e * HDIM] = f2bf(acc[t][e] + bv);
    }
}

// =====================================================================
// Kernel 2: fused flash attention, block-diagonal (varlen) mask
//   K block : TDM tensor_load_to_lds (TENSORcnt), LDS-padded rows via D# pad
//   V block : global_load_async_to_lds_b128 (ASYNCcnt) + in-LDS transpose
// =====================================================================
__global__ __launch_bounds__(128) void attn_fused(
        const unsigned short* __restrict__ q_ws, const unsigned short* __restrict__ k_ws,
        const unsigned short* __restrict__ v_ws, const int* __restrict__ seg,
        unsigned short* __restrict__ o_ws) {
    __shared__ unsigned short kbuf[32][72];       // [key][d], TDM pads 128B+16B rows
    __shared__ unsigned short vstage[32][72];     // [key][d] async-staged
    __shared__ unsigned short vbuf[64][40];       // [d][key] transposed
    __shared__ unsigned short pbuf[4][16][40];    // per-wave P transpose scratch

    const int head = blockIdx.y;
    const int q0   = blockIdx.x * 64;
    const int tid  = threadIdx.x, wave = tid >> 5, lane = tid & 31;
    const int half = lane >> 4, l16 = lane & 15;
    const int qbase = q0 + wave * 16;

    const unsigned short* qh = q_ws + (size_t)head * S_TOT * HDIM;
    const unsigned short* kh = k_ws + (size_t)head * S_TOT * HDIM;
    const unsigned short* vh = v_ws + (size_t)head * S_TOT * HDIM;

    const unsigned short* qrow = qh + (size_t)(qbase + l16) * HDIM;
    v16u qa0 = ld2(qrow +      half * 8, qrow + 16 + half * 8);
    v16u qa1 = ld2(qrow + 32 + half * 8, qrow + 48 + half * 8);

    int seg_q[8];
    #pragma unroll
    for (int e = 0; e < 8; ++e) seg_q[e] = seg[qbase + e + half * 8];

    float m_run[8], l_run[8];
    v8f o[4] = {};
    #pragma unroll
    for (int e = 0; e < 8; ++e) { m_run[e] = -3.0e38f; l_run[e] = 0.f; }

    const float scale = 0.125f;

    // ---- TDM descriptor (uniform, SGPR-resident) ----
    // group1: data_size=2B(code1), pad_enable, pad_interval=32dw(code4),
    //         pad_amount=4dw(code3); tensor 64x4096, tile 64x32, stride 64
    const unsigned kbase = (unsigned)(size_t)&kbuf[0][0];
    i32x8 g1;
    g1[0] = (int)((1u << 16) | (1u << 20) | (4u << 22) | (3u << 25));
    g1[1] = (int)((unsigned)HDIM << 16);          // tensor_dim0 = 64
    g1[2] = (int)((unsigned)S_TOT << 16);         // tensor_dim1 = 4096
    g1[3] = (int)((unsigned)HDIM << 16);          // tile_dim0 = 64
    g1[4] = 32;                                   // tile_dim1 = 32
    g1[5] = HDIM;                                 // tensor_dim0_stride = 64
    g1[6] = 0;
    g1[7] = 0;
    const i32x4 gz4 = {};
    const i32x8 gz8 = {};

    // V async staging addresses (per-lane)
    const int r  = tid >> 2;           // 0..31 local key
    const int c8 = (tid & 3) * 16;     // 0,16,32,48 (d)
    const unsigned vlds0 = (unsigned)(size_t)&vstage[r][c8];
    const unsigned vlds1 = vlds0 + 16;

    for (int j = 0; j < S_TOT; j += 32) {
        // --- K block via Tensor Data Mover (one DMA per block) ---
        if (wave == 0) {
            unsigned long long ga = (unsigned long long)(size_t)(kh + (size_t)j * HDIM);
            u32x4 g0;
            g0[0] = 1u;                                   // count=1
            g0[1] = kbase;                                // lds_addr
            g0[2] = (unsigned)ga;                         // global_addr lo
            g0[3] = ((unsigned)(ga >> 32) & 0x01FFFFFFu) | (2u << 30); // hi | type=2
            __builtin_amdgcn_tensor_load_to_lds(g0, g1, gz4, gz4, gz8, 0);
        }
        // --- V block via async global->LDS (ASYNCcnt) ---
        {
            unsigned long long vg = (unsigned long long)(size_t)
                (vh + (size_t)(j + r) * HDIM + c8);
            asm volatile("global_load_async_to_lds_b128 %0, %1, off"
                         :: "v"(vlds0), "v"(vg) : "memory");
            asm volatile("global_load_async_to_lds_b128 %0, %1, off offset:16"
                         :: "v"(vlds1), "v"(vg) : "memory");
        }
        if (j + 32 < S_TOT) {
            __builtin_prefetch(kh + (size_t)(j + 32 + r) * HDIM + c8, 0, 1);
            __builtin_prefetch(vh + (size_t)(j + 32 + r) * HDIM + c8, 0, 1);
        }
        // --- transpose V inside LDS once async data lands ---
        asm volatile("s_wait_asynccnt 0" ::: "memory");
        {
            v8u v0 = *(const v8u*)&vstage[r][c8];
            v8u v1 = *(const v8u*)&vstage[r][c8 + 8];
            #pragma unroll
            for (int c = 0; c < 8; ++c) {
                vbuf[c8 + c][r]     = v0[c];
                vbuf[c8 + 8 + c][r] = v1[c];
            }
        }
        if (wave == 0) __builtin_amdgcn_s_wait_tensorcnt(0);
        __syncthreads();

        // --- scores: S = Q @ K^T, two 16x16 key tiles ---
        v8f s0 = {}, s1 = {};
        {
            const unsigned short* kr0 = &kbuf[l16][0];
            const unsigned short* kr1 = &kbuf[16 + l16][0];
            v16u b00 = ld2(kr0 +      half * 8, kr0 + 16 + half * 8);
            v16u b01 = ld2(kr0 + 32 + half * 8, kr0 + 48 + half * 8);
            v16u b10 = ld2(kr1 +      half * 8, kr1 + 16 + half * 8);
            v16u b11 = ld2(kr1 + 32 + half * 8, kr1 + 48 + half * 8);
            s0 = wmma_bf16(qa0, b00, s0);
            s0 = wmma_bf16(qa1, b01, s0);
            s1 = wmma_bf16(qa0, b10, s1);
            s1 = wmma_bf16(qa1, b11, s1);
        }

        // --- mask + online softmax ---
        int segk0 = seg[j + l16];
        int segk1 = seg[j + 16 + l16];
        float p0[8], p1[8], mnew[8], alpha[8];
        #pragma unroll
        for (int e = 0; e < 8; ++e) {
            float a = (seg_q[e] == segk0) ? s0[e] * scale : -1.0e9f;
            float b = (seg_q[e] == segk1) ? s1[e] * scale : -1.0e9f;
            p0[e] = a; p1[e] = b;
            float mx = fmaxf(a, b);
            mx = fmaxf(mx, __shfl_xor(mx, 1, 32));
            mx = fmaxf(mx, __shfl_xor(mx, 2, 32));
            mx = fmaxf(mx, __shfl_xor(mx, 4, 32));
            mx = fmaxf(mx, __shfl_xor(mx, 8, 32));
            mnew[e] = fmaxf(m_run[e], mx);
        }
        #pragma unroll
        for (int e = 0; e < 8; ++e) {
            alpha[e] = __expf(m_run[e] - mnew[e]);
            p0[e] = __expf(p0[e] - mnew[e]);
            p1[e] = __expf(p1[e] - mnew[e]);
            float rs = p0[e] + p1[e];
            rs += __shfl_xor(rs, 1, 32);
            rs += __shfl_xor(rs, 2, 32);
            rs += __shfl_xor(rs, 4, 32);
            rs += __shfl_xor(rs, 8, 32);
            l_run[e] = l_run[e] * alpha[e] + rs;
            m_run[e] = mnew[e];
        }
        #pragma unroll
        for (int t = 0; t < 4; ++t)
            #pragma unroll
            for (int e = 0; e < 8; ++e) o[t][e] *= alpha[e];

        // --- transpose P through per-wave LDS scratch ---
        unsigned short (*pb)[40] = pbuf[wave];
        #pragma unroll
        for (int e = 0; e < 8; ++e) {
            pb[e + half * 8][l16]      = f2bf(p0[e]);
            pb[e + half * 8][16 + l16] = f2bf(p1[e]);
        }
        asm volatile("s_wait_dscnt 0" ::: "memory");

        const unsigned short* prow = &pb[l16][0];
        v16u pa = ld2(prow + half * 8, prow + 16 + half * 8);

        // --- O += P @ V ---
        v16u vfr[4];
        #pragma unroll
        for (int t = 0; t < 4; ++t) {
            const unsigned short* vr = &vbuf[t * 16 + l16][0];
            vfr[t] = ld2(vr + half * 8, vr + 16 + half * 8);
        }
        #pragma unroll
        for (int t = 0; t < 4; ++t) o[t] = wmma_bf16(pa, vfr[t], o[t]);
        __syncthreads();
    }

    // normalize + store bf16 token-major [S][1024]
    #pragma unroll
    for (int t = 0; t < 4; ++t) {
        unsigned short* base = o_ws +
            (size_t)(qbase + half * 8) * DIMK + head * HDIM + t * 16 + l16;
        #pragma unroll
        for (int e = 0; e < 8; ++e)
            base[(size_t)e * DIMK] = f2bf(o[t][e] / l_run[e]);
    }
}

// =====================================================================
// Kernel 3: out = attn_out(bf16) @ w_proj(bf16) + b_proj  -> fp32 d_out
// =====================================================================
__global__ __launch_bounds__(256) void proj_gemm(
        const unsigned short* __restrict__ a, const unsigned short* __restrict__ wb,
        const float* __restrict__ bias, float* __restrict__ out) {
    __shared__ unsigned short alds[64][40];
    __shared__ unsigned short blds[128][40];

    const int tid  = threadIdx.x;
    const int wave = tid >> 5, lane = tid & 31;
    const int wm   = wave & 3, wn = wave >> 2;
    const int half = lane >> 4, l16 = lane & 15;
    const int m0   = blockIdx.x * 64;
    const int n0   = blockIdx.y * 128;

    v8f acc[4] = {};

    for (int k0 = 0; k0 < DIMK; k0 += 32) {
        {
            int r  = tid >> 2;
            int c8 = (tid & 3) * 8;
            *(v8u*)&alds[r][c8] = *(const v8u*)(a + (size_t)(m0 + r) * DIMK + k0 + c8);
        }
        {
            int r  = tid >> 4;
            int c8 = (tid & 15) * 8;
            #pragma unroll
            for (int rr = 0; rr < 32; rr += 16) {
                v8u v = *(const v8u*)(wb + (size_t)(k0 + r + rr) * DIMK + n0 + c8);
                #pragma unroll
                for (int c = 0; c < 8; ++c) blds[c8 + c][r + rr] = v[c];
            }
        }
        __syncthreads();

        const unsigned short* arow = &alds[wm * 16 + l16][0];
        v16u af = ld2(arow + half * 8, arow + 16 + half * 8);
        v16u bfr[4];
        #pragma unroll
        for (int t = 0; t < 4; ++t) {
            const unsigned short* brow = &blds[wn * 64 + t * 16 + l16][0];
            bfr[t] = ld2(brow + half * 8, brow + 16 + half * 8);
        }
        #pragma unroll
        for (int t = 0; t < 4; ++t) acc[t] = wmma_bf16(af, bfr[t], acc[t]);
        __syncthreads();
    }

    #pragma unroll
    for (int t = 0; t < 4; ++t) {
        int n = n0 + wn * 64 + t * 16 + l16;
        float bv = bias[n];
        float* base = out + (size_t)(m0 + wm * 16 + half * 8) * DIMK + n;
        #pragma unroll
        for (int e = 0; e < 8; ++e)
            base[(size_t)e * DIMK] = acc[t][e] + bv;
    }
}

// =====================================================================
extern "C" void kernel_launch(void* const* d_in, const int* in_sizes, int n_in,
                              void* d_out, int out_size, void* d_ws, size_t ws_size,
                              hipStream_t stream) {
    const float* x_in   = (const float*)d_in[0];
    const float* w_qkv  = (const float*)d_in[1];
    const float* b_qkv  = (const float*)d_in[2];
    const float* w_proj = (const float*)d_in[3];
    const float* b_proj = (const float*)d_in[4];
    const int*   seg    = (const int*)d_in[5];

    const size_t plane = (size_t)S_TOT * DIMK;
    unsigned short* q_ws = (unsigned short*)d_ws;
    unsigned short* k_ws = q_ws + plane;
    unsigned short* v_ws = k_ws + plane;
    unsigned short* o_ws = v_ws + plane;
    unsigned short* xb   = o_ws + plane;
    unsigned short* wqb  = xb + plane;
    unsigned short* wpb  = wqb + (size_t)DIMK * N_QKV;

    cvt_bf16<<<(int)(plane / 4 / 256), 256, 0, stream>>>(
        (const float4*)x_in, (uint2*)xb, (int)(plane / 4));
    cvt_bf16<<<(int)((size_t)DIMK * N_QKV / 4 / 256), 256, 0, stream>>>(
        (const float4*)w_qkv, (uint2*)wqb, (int)((size_t)DIMK * N_QKV / 4));
    cvt_bf16<<<(int)((size_t)DIMK * DIMK / 4 / 256), 256, 0, stream>>>(
        (const float4*)w_proj, (uint2*)wpb, (int)((size_t)DIMK * DIMK / 4));

    qkv_gemm<<<dim3(S_TOT / 64, N_QKV / 128), 256, 0, stream>>>(
        xb, wqb, b_qkv, q_ws, k_ws, v_ws);
    attn_fused<<<dim3(S_TOT / 64, NHEAD), 128, 0, stream>>>(
        q_ws, k_ws, v_ws, seg, o_ws);
    proj_gemm<<<dim3(S_TOT / 64, DIMK / 128), 256, 0, stream>>>(
        o_ws, wpb, b_proj, (float*)d_out);
}
